// GNNRefinement_46119358825302
// MI455X (gfx1250) — matrix operand order repaired and checked by reading, not compiled, not run on hardware
//
#include <hip/hip_runtime.h>
#include <hip/hip_bf16.h>
#include <math.h>

// Shapes from reference: B=256, IN_FEAT=7, U=32, A=64, F=32, OUT=4
#define PIX (256 * 32 * 64)   // 524288 pixels
#define UA  (32 * 64)         // 2048

typedef __attribute__((ext_vector_type(16))) _Float16 v16h;
typedef __attribute__((ext_vector_type(8)))  _Float16 v8h;
typedef __attribute__((ext_vector_type(8)))  float    v8f;

// ---------------------------------------------------------------------------
// Weight conversion: fp32 [rows][cols] -> f16 [rowsPad][kpad], zero padded
// ---------------------------------------------------------------------------
__global__ __launch_bounds__(256)
void cvt_w(const float* __restrict__ w, _Float16* __restrict__ o,
           int rows, int cols, int rowsPad, int kpad) {
  int i = blockIdx.x * 256 + threadIdx.x;
  int tot = rowsPad * kpad;
  if (i >= tot) return;
  int r = i / kpad, c = i % kpad;
  o[i] = (r < rows && c < cols) ? (_Float16)w[r * cols + c] : (_Float16)0.f;
}

// ---------------------------------------------------------------------------
// Pack kernels: build f16 activation matrices [pixel][Kpad]
// pixel p = (b*U + u)*A + a ;  b = p>>11, r = p&2047
// ---------------------------------------------------------------------------
__global__ __launch_bounds__(256)
void pack_cgcg(const float* __restrict__ cg, _Float16* __restrict__ X) {
  int p = blockIdx.x * 256 + threadIdx.x;
  if (p >= PIX) return;
  int b = p >> 11, r = p & 2047;
  const float* cgp = cg + (size_t)b * 7 * UA + r;
  _Float16* row = X + (size_t)p * 32;
  for (int c = 0; c < 7; c++) {
    _Float16 v = (_Float16)cgp[(size_t)c * UA];
    row[c] = v; row[7 + c] = v;
  }
  for (int c = 14; c < 32; c++) row[c] = (_Float16)0.f;
}

__global__ __launch_bounds__(256)
void pack_cg_s(const float* __restrict__ cg, const _Float16* __restrict__ S,
               _Float16* __restrict__ X) {
  int p = blockIdx.x * 256 + threadIdx.x;
  if (p >= PIX) return;
  int b = p >> 11, r = p & 2047;
  const float* cgp = cg + (size_t)b * 7 * UA + r;
  _Float16* row = X + (size_t)p * 96;
  for (int c = 0; c < 7; c++) row[c] = (_Float16)cgp[(size_t)c * UA];
  const _Float16* srow = S + (size_t)p * 64;
  for (int c = 0; c < 64; c++) row[7 + c] = srow[c];
  for (int c = 71; c < 96; c++) row[c] = (_Float16)0.f;
}

__global__ __launch_bounds__(256)
void pack_rr(const _Float16* __restrict__ R, const _Float16* __restrict__ S,
             _Float16* __restrict__ X) {
  int p = blockIdx.x * 256 + threadIdx.x;
  if (p >= PIX) return;
  _Float16* row = X + (size_t)p * 128;
  const _Float16* rrow = R + (size_t)p * 64;
  const _Float16* srow = S + (size_t)p * 64;
  for (int c = 0; c < 64; c++) { row[c] = rrow[c]; row[64 + c] = srow[c]; }
}

// ---------------------------------------------------------------------------
// pp: out[:, :32] = in[:, :32] ; out[:,32+j,u,a] = (sum_v in[:,32+j,v,a] - in)/31
// optional relu applied to input first. Block per (b,a): 256*64 blocks.
// ---------------------------------------------------------------------------
__global__ __launch_bounds__(256)
void pp_kernel(const _Float16* __restrict__ I, _Float16* __restrict__ O, int doRelu) {
  __shared__ float part[8][32];
  __shared__ float csum[32];
  int blk = blockIdx.x;
  int b = blk >> 6, a = blk & 63;
  size_t base = ((size_t)b * 32) * 64 + a;   // pixel index at u=0
  int t = threadIdx.x, j = t & 31, g = t >> 5;
  float s = 0.f;
  for (int u = g; u < 32; u += 8) {
    size_t pix = base + (size_t)u * 64;
    float v = (float)I[pix * 64 + 32 + j];
    if (doRelu) v = fmaxf(v, 0.f);
    s += v;
  }
  part[g][j] = s;
  __syncthreads();
  if (g == 0) {
    float acc = 0.f;
    for (int gg = 0; gg < 8; gg++) acc += part[gg][j];
    csum[j] = acc;
  }
  __syncthreads();
  for (int idx = t; idx < 2048; idx += 256) {
    int u = idx >> 6, c = idx & 63;
    size_t pix = base + (size_t)u * 64;
    float v = (float)I[pix * 64 + c];
    if (doRelu) v = fmaxf(v, 0.f);
    float o = (c < 32) ? v : (csum[c - 32] - v) * (1.0f / 31.0f);
    O[pix * 64 + c] = (_Float16)o;
  }
}

// ---------------------------------------------------------------------------
// agg: out[:,c,u,a] = (sum_k in[:,c,u,k] - in[:,c,u,a]) / 63. Block per (b,u).
// ---------------------------------------------------------------------------
__global__ __launch_bounds__(256)
void agg_kernel(const _Float16* __restrict__ I, _Float16* __restrict__ O) {
  __shared__ float part[4][64];
  __shared__ float csum[64];
  int blk = blockIdx.x;
  int b = blk >> 5, u = blk & 31;
  size_t base = ((size_t)b * 32 + u) * 64;   // pixel at a=0
  int t = threadIdx.x, c = t & 63, g = t >> 6;
  float s = 0.f;
  for (int a = g; a < 64; a += 4) s += (float)I[(base + a) * 64 + c];
  part[g][c] = s;
  __syncthreads();
  if (g == 0) {
    float acc = 0.f;
    for (int gg = 0; gg < 4; gg++) acc += part[gg][c];
    csum[c] = acc;
  }
  __syncthreads();
  for (int idx = t; idx < 4096; idx += 256) {
    int a = idx >> 6, cc = idx & 63;
    size_t pix = base + a;
    float v = (float)I[pix * 64 + cc];
    O[pix * 64 + cc] = (_Float16)((csum[cc] - v) * (1.0f / 63.0f));
  }
}

// ---------------------------------------------------------------------------
// WMMA GEMM: Y[p][0..63] = relu?(W[64][KPAD] x X[KPAD][p] + bias)
// Wave tile: M=64 x N=32 pixels (2 N-tiles, 4 M-tiles, 8 accumulators).
// v_wmma_f32_16x16x32_f16, f32 accumulate.
// A layout (16x32 f16): lane L row M=L%16, e0..7 <- K=(L/16)*16+e,
//                       e8..15 <- K=(L/16)*16+e   (two 16B loads)
// B layout (32x16 f16): lane L col N=L%16, e <- K=(L/16)*16+e (one 32B load)
// D layout: lane L, vgpr v <-> (M = v + 8*(L/16), N = L%16)
// ---------------------------------------------------------------------------
template <int KPAD>
__global__ __launch_bounds__(256)
void gemm_act(const _Float16* __restrict__ X, const _Float16* __restrict__ W,
              const float* __restrict__ bias, _Float16* __restrict__ Y, int doRelu) {
  const int lane = threadIdx.x & 31;
  const int wave = threadIdx.x >> 5;
  const int p0 = (blockIdx.x * 8 + wave) * 32;   // 32 pixels per wave
  const int ln = lane & 15, hi = lane >> 4;

  const _Float16* xrow0 = X + (size_t)(p0 + ln) * KPAD + hi * 16;
  const _Float16* xrow1 = xrow0 + (size_t)16 * KPAD;
  const _Float16* wrow  = W + (size_t)ln * KPAD + hi * 16;

  v8f z = {0.f, 0.f, 0.f, 0.f, 0.f, 0.f, 0.f, 0.f};
  v8f acc[4][2];
#pragma unroll
  for (int mt = 0; mt < 4; ++mt) { acc[mt][0] = z; acc[mt][1] = z; }

#pragma unroll
  for (int k = 0; k < KPAD; k += 32) {
    v16h bmat0 = *(const v16h*)(xrow0 + k);
    v16h bmat1 = *(const v16h*)(xrow1 + k);
#pragma unroll
    for (int mt = 0; mt < 4; ++mt) {
      const _Float16* wr = wrow + (size_t)mt * 16 * KPAD + k;
      v8h alo = *(const v8h*)(wr);
      v8h ahi = *(const v8h*)(wr + 8);
      v16h amat;
#pragma unroll
      for (int e = 0; e < 8; e++) { amat[e] = alo[e]; amat[8 + e] = ahi[e]; }
      acc[mt][0] = __builtin_amdgcn_wmma_f32_16x16x32_f16(
          false, amat, false, bmat0, (short)0, acc[mt][0], false, false);
      acc[mt][1] = __builtin_amdgcn_wmma_f32_16x16x32_f16(
          false, amat, false, bmat1, (short)0, acc[mt][1], false, false);
    }
  }

#pragma unroll
  for (int nt = 0; nt < 2; ++nt) {
    _Float16* yrow = Y + (size_t)(p0 + nt * 16 + ln) * 64;
#pragma unroll
    for (int mt = 0; mt < 4; ++mt) {
#pragma unroll
      for (int v = 0; v < 8; ++v) {
        int c = mt * 16 + hi * 8 + v;
        float val = acc[mt][nt][v] + bias[c];
        if (doRelu) val = fmaxf(val, 0.f);
        yrow[c] = (_Float16)val;
      }
    }
  }
}

// Final layer: 4 out channels (weights zero-padded to M=16), K=64,
// fused softplus(x)/10 (hardware transcendentals), fp32 out (B, 4, U, A).
__global__ __launch_bounds__(256)
void gemm_final(const _Float16* __restrict__ X, const _Float16* __restrict__ W,
                const float* __restrict__ bias, float* __restrict__ out) {
  const int lane = threadIdx.x & 31;
  const int wave = threadIdx.x >> 5;
  const int p0 = (blockIdx.x * 8 + wave) * 16;
  const int ln = lane & 15, hi = lane >> 4;

  const _Float16* xrow = X + (size_t)(p0 + ln) * 64 + hi * 16;
  const _Float16* wrow = W + (size_t)ln * 64 + hi * 16;

  v8f acc = {0.f, 0.f, 0.f, 0.f, 0.f, 0.f, 0.f, 0.f};
#pragma unroll
  for (int k = 0; k < 64; k += 32) {
    v16h bmat = *(const v16h*)(xrow + k);
    v8h alo = *(const v8h*)(wrow + k);
    v8h ahi = *(const v8h*)(wrow + k + 8);
    v16h amat;
#pragma unroll
    for (int e = 0; e < 8; e++) { amat[e] = alo[e]; amat[8 + e] = ahi[e]; }
    acc = __builtin_amdgcn_wmma_f32_16x16x32_f16(
        false, amat, false, bmat, (short)0, acc, false, false);
  }

  if (hi == 0) {   // only M=0..3 are real output channels
    int p = p0 + ln;
    int b = p >> 11, r = p & 2047;
#pragma unroll
    for (int v = 0; v < 4; ++v) {
      float val = acc[v] + bias[v];
      // stable softplus via v_exp_f32 / v_log_f32 fast paths
      float sp = (val > 20.f) ? val : __logf(1.f + __expf(val));
      out[((size_t)b * 4 + v) * UA + r] = sp * 0.1f;
    }
  }
}

// ---------------------------------------------------------------------------
// Host launcher
// ---------------------------------------------------------------------------
extern "C" void kernel_launch(void* const* d_in, const int* in_sizes, int n_in,
                              void* d_out, int out_size, void* d_ws, size_t ws_size,
                              hipStream_t stream) {
  const float* cg   = (const float*)d_in[0];
  const float* p1w1 = (const float*)d_in[1];
  const float* p1b1 = (const float*)d_in[2];
  const float* p1w2 = (const float*)d_in[3];
  const float* p1b2 = (const float*)d_in[4];
  const float* p2w1 = (const float*)d_in[5];
  const float* p2b1 = (const float*)d_in[6];
  const float* p2w2 = (const float*)d_in[7];
  const float* p2b2 = (const float*)d_in[8];
  const float* g1w1 = (const float*)d_in[9];
  const float* g1b1 = (const float*)d_in[10];
  const float* g1w2 = (const float*)d_in[11];
  const float* g1b2 = (const float*)d_in[12];
  const float* g2w1 = (const float*)d_in[13];
  const float* g2b1 = (const float*)d_in[14];
  const float* g2w2 = (const float*)d_in[15];
  const float* g2b2 = (const float*)d_in[16];

  size_t off = 0;
  auto carve = [&](size_t elems) -> _Float16* {
    _Float16* p = (_Float16*)((char*)d_ws + off);
    off += ((elems * sizeof(_Float16)) + 255) & ~(size_t)255;
    return p;
  };

  _Float16* XB   = carve((size_t)PIX * 128);  // staging, reused at pad 32/96/128
  _Float16* BufA = carve((size_t)PIX * 64);
  _Float16* BufB = carve((size_t)PIX * 64);
  _Float16* BufC = carve((size_t)PIX * 64);
  _Float16* Wp1a = carve(64 * 32);
  _Float16* Wp1b = carve(64 * 64);
  _Float16* Wp2a = carve(64 * 96);
  _Float16* Wp2b = carve(64 * 64);
  _Float16* Wg1a = carve(64 * 96);
  _Float16* Wg1b = carve(64 * 64);
  _Float16* Wg2a = carve(64 * 128);
  _Float16* Wg2b = carve(16 * 64);
  (void)ws_size; (void)in_sizes; (void)n_in; (void)out_size;

  const int packGrid  = PIX / 256;        // 2048
  const int gemmGrid  = PIX / 32 / 8;     // 2048 (8 waves/block, 32 px/wave)
  const int finalGrid = PIX / 16 / 8;     // 4096 (16 px/wave)
  const int ppGrid    = 256 * 64;         // 16384  (b,a)
  const int aggGrid   = 256 * 32;         // 8192   (b,u)

  // f16 weight staging (zero padded)
  cvt_w<<<(64 * 32 + 255) / 256, 256, 0, stream>>>(p1w1, Wp1a, 64, 14, 64, 32);
  cvt_w<<<(64 * 64 + 255) / 256, 256, 0, stream>>>(p1w2, Wp1b, 64, 64, 64, 64);
  cvt_w<<<(64 * 96 + 255) / 256, 256, 0, stream>>>(p2w1, Wp2a, 64, 71, 64, 96);
  cvt_w<<<(64 * 64 + 255) / 256, 256, 0, stream>>>(p2w2, Wp2b, 64, 64, 64, 64);
  cvt_w<<<(64 * 96 + 255) / 256, 256, 0, stream>>>(g1w1, Wg1a, 64, 71, 64, 96);
  cvt_w<<<(64 * 64 + 255) / 256, 256, 0, stream>>>(g1w2, Wg1b, 64, 64, 64, 64);
  cvt_w<<<(64 * 128 + 255) / 256, 256, 0, stream>>>(g2w1, Wg2a, 64, 128, 64, 128);
  cvt_w<<<(16 * 64 + 255) / 256, 256, 0, stream>>>(g2w2, Wg2b, 4, 64, 16, 64);

  // phi 1: msgs = pp(conv2(pp(relu(conv1([cg,cg])))))
  pack_cgcg<<<packGrid, 256, 0, stream>>>(cg, XB);
  gemm_act<32><<<gemmGrid, 256, 0, stream>>>(XB, Wp1a, p1b1, BufA, 1);
  pp_kernel<<<ppGrid, 256, 0, stream>>>(BufA, BufB, 0);
  gemm_act<64><<<gemmGrid, 256, 0, stream>>>(BufB, Wp1b, p1b2, BufA, 0);
  pp_kernel<<<ppGrid, 256, 0, stream>>>(BufA, BufB, 0);          // msgs
  agg_kernel<<<aggGrid, 256, 0, stream>>>(BufB, BufA);           // agg(msgs)

  // gamma 1 (non-final): r
  pack_cg_s<<<packGrid, 256, 0, stream>>>(cg, BufA, XB);
  gemm_act<96><<<gemmGrid, 256, 0, stream>>>(XB, Wg1a, g1b1, BufA, 1);
  pp_kernel<<<ppGrid, 256, 0, stream>>>(BufA, BufB, 0);
  gemm_act<64><<<gemmGrid, 256, 0, stream>>>(BufB, Wg1b, g1b2, BufA, 0);
  pp_kernel<<<ppGrid, 256, 0, stream>>>(BufA, BufB, 1);          // r = pp(relu(.))

  // phi 2: msgs = phi([cg, r])   (r stays live in BufB)
  pack_cg_s<<<packGrid, 256, 0, stream>>>(cg, BufB, XB);
  gemm_act<96><<<gemmGrid, 256, 0, stream>>>(XB, Wp2a, p2b1, BufA, 1);
  pp_kernel<<<ppGrid, 256, 0, stream>>>(BufA, BufC, 0);
  gemm_act<64><<<gemmGrid, 256, 0, stream>>>(BufC, Wp2b, p2b2, BufA, 0);
  pp_kernel<<<ppGrid, 256, 0, stream>>>(BufA, BufC, 0);          // msgs2
  agg_kernel<<<aggGrid, 256, 0, stream>>>(BufC, BufA);           // agg(msgs2)

  // gamma 2 (final): softplus(conv2(pp(relu(conv1([r, agg])))))/10
  pack_rr<<<packGrid, 256, 0, stream>>>(BufB, BufA, XB);
  gemm_act<128><<<gemmGrid, 256, 0, stream>>>(XB, Wg2a, g2b1, BufA, 1);
  pp_kernel<<<ppGrid, 256, 0, stream>>>(BufA, BufC, 0);
  gemm_final<<<finalGrid, 256, 0, stream>>>(BufC, Wg2b, g2b2, (float*)d_out);
}
